// FraudDetectorGAT_87445534147094
// MI455X (gfx1250) — matrix operand order repaired and checked by reading, not compiled, not run on hardware
//
#include <hip/hip_runtime.h>
#include <hip/hip_bf16.h>

typedef __attribute__((ext_vector_type(16))) _Float16 v16h;
typedef __attribute__((ext_vector_type(2)))  _Float16 v2h;
typedef __attribute__((ext_vector_type(8)))  float    v8f;

#define T_TYPES   11
#define N_ENT     100000
#define N_TX      100000
#define DIM       128
#define E_EDGES   500000
#define F_TX      394
#define NEG_SLOPE 0.2f
#define LN_EPS    1e-5f

#define BM 128
#define BK 32

// ---------------------------------------------------------------- utilities

__global__ void k_fill(float* __restrict__ p, float v, int n) {
  int i = blockIdx.x * blockDim.x + threadIdx.x;
  if (i < n) p[i] = v;
}

__global__ void k_relu(float* __restrict__ p, int n) {
  int i = blockIdx.x * blockDim.x + threadIdx.x;
  if (i < n) p[i] = fmaxf(p[i], 0.f);
}

__device__ __forceinline__ float lrelu(float v) { return v > 0.f ? v : NEG_SLOPE * v; }

__device__ __forceinline__ void atomicMaxF(float* addr, float val) {
  unsigned int* ua = (unsigned int*)addr;
  unsigned int cur = *ua;
  while (__uint_as_float(cur) < val) {
    unsigned int prev = atomicCAS(ua, cur, __float_as_uint(val));
    if (prev == cur) break;
    cur = prev;
  }
}

// Pack a K x N f32 row-major matrix (row stride == N) into WMMA B-fragment
// order: [kstep][jtile][lane][i] halves, each lane's 16 halves contiguous.
// K zero-padded to a multiple of 32. total = ceil(K/32)*32*N.
__global__ void k_pack_bfrag(const float* __restrict__ src, _Float16* __restrict__ dst,
                             int K, int N, int total)
{
  int d = blockIdx.x * blockDim.x + threadIdx.x;
  if (d >= total) return;
  int perstep = N * 32;            // halves per k-step
  int kstep = d / perstep;
  int rem   = d % perstep;
  int j = rem >> 9;                // 32*16 = 512 halves per 16-col tile
  int l = (rem >> 4) & 31;
  int i = rem & 15;
  int k = kstep * 32 + ((l >> 4) << 4) + i;   // K = 16*hi + i
  int n = j * 16 + (l & 15);
  dst[d] = (k < K) ? (_Float16)src[(long)k * N + n] : (_Float16)0.f;
}

// ---------------------------------------------------------------- WMMA GEMM
// C[M x (NT*16)] = gather(A)[M x K] @ Bf  (Bf pre-packed in fragment order).
// A tiles double-buffered in LDS fragment layout with register-staged
// prefetch; B fragments hoisted into a register array so all NT loads issue
// in one clause and the NT WMMAs run back-to-back.
template <int NT>
__global__ __launch_bounds__(256) void k_gemm_wmma(
    const float* __restrict__ A, const int* __restrict__ gidx,
    const _Float16* __restrict__ Bf, float* __restrict__ C,
    const float* __restrict__ bias, int M, int K, int lda, int relu)
{
  constexpr int N = NT * 16;
  __shared__ __align__(32) _Float16 AsF[2][8][32][16];  // [buf][strip][lane][i]

  const int tid  = threadIdx.x;
  const int lane = tid & 31;
  const int wv   = tid >> 5;           // 8 waves, one 16-row strip each
  const int row0 = blockIdx.x * BM;

  // Per-thread staging geometry (constant across K-steps, since 256 % 16 == 0):
  const int kk   = (tid & 15) << 1;              // K-pair within a 32-wide step
  const int base = tid >> 4;                     // row within each 16-row strip
  const int shi  = kk >> 4;                      // fragment half-wave
  const int sii  = kk & 15;                      // fragment element (even)
  const int slot = (base & 15) | (shi << 4);

  // Resolve gather rows / base pointers once (8 strips per thread).
  const float* abase[8];
  bool rvalid[8];
#pragma unroll
  for (int u = 0; u < 8; ++u) {
    int gr = row0 + base + u * 16;
    rvalid[u] = gr < M;
    int sr = rvalid[u] ? (gidx ? gidx[gr] : gr) : 0;
    abase[u] = A + (long)sr * lda;
  }

  v8f acc[NT];
#pragma unroll
  for (int j = 0; j < NT; ++j)
#pragma unroll
    for (int e = 0; e < 8; ++e) acc[j][e] = 0.f;

  // Prefetch first A tile into registers.
  float v0r[8], v1r[8];
#pragma unroll
  for (int u = 0; u < 8; ++u) {
    float v0 = 0.f, v1 = 0.f;
    if (rvalid[u] && kk < K) {
      v0 = abase[u][kk];
      if (kk + 1 < K) v1 = abase[u][kk + 1];
    }
    v0r[u] = v0; v1r[u] = v1;
  }

  const int nsteps = (K + BK - 1) / BK;
  for (int step = 0; step < nsteps; ++step) {
    const int buf = step & 1;
    // commit staged registers to LDS in fragment layout (packed b32 stores)
#pragma unroll
    for (int u = 0; u < 8; ++u) {
      v2h pk; pk[0] = (_Float16)v0r[u]; pk[1] = (_Float16)v1r[u];
      *(v2h*)&AsF[buf][u][slot][sii] = pk;
    }
    __syncthreads();

    // issue next tile's global loads; they overlap the WMMAs below
    if (step + 1 < nsteps) {
      int k0 = (step + 1) * BK;
#pragma unroll
      for (int u = 0; u < 8; ++u) {
        float v0 = 0.f, v1 = 0.f;
        if (rvalid[u] && (k0 + kk) < K) {
          v0 = abase[u][k0 + kk];
          if (k0 + kk + 1 < K) v1 = abase[u][k0 + kk + 1];
        }
        v0r[u] = v0; v1r[u] = v1;
      }
    }

    v16h afrag = *(const v16h*)(&AsF[buf][wv][lane][0]);
    const _Float16* bb = Bf + (size_t)step * (NT * 512) + (size_t)lane * 16;
    v16h bfr[NT];
#pragma unroll
    for (int j = 0; j < NT; ++j)
      bfr[j] = *(const v16h*)(bb + j * 512);
#pragma unroll
    for (int j = 0; j < NT; ++j)
      acc[j] = __builtin_amdgcn_wmma_f32_16x16x32_f16(
          false, afrag, false, bfr[j], (short)0, acc[j], false, false);
    // no trailing barrier: double buffer gives a 2-step WAR distance and
    // __syncthreads drains DS before signaling.
  }

  // C layout: VGPR r -> M = r + 8*(lane>=16) ; N = lane&15
  const int hi = lane >> 4;
#pragma unroll
  for (int j = 0; j < NT; ++j) {
    int col = j * 16 + (lane & 15);
#pragma unroll
    for (int r = 0; r < 8; ++r) {
      int grow = row0 + wv * 16 + r + (hi ? 8 : 0);
      if (grow < M) {
        float v = acc[j][r];
        if (bias) v += bias[col];
        if (relu) v = fmaxf(v, 0.f);
        C[(long)grow * N + col] = v;
      }
    }
  }
}

// ---------------------------------------------------------------- GAT pieces

__global__ void k_att_dots(const float* __restrict__ xw,
                           const float* __restrict__ att_s,
                           const float* __restrict__ att_d,
                           float* __restrict__ a_s, float* __restrict__ a_d, int M)
{
  int gw = (blockIdx.x * blockDim.x + threadIdx.x) >> 5;
  int lane = threadIdx.x & 31;
  if (gw >= M) return;
  const float* row = xw + (long)gw * DIM;
  float ss = 0.f, sd = 0.f;
#pragma unroll
  for (int i = 0; i < 4; ++i) {
    int c = i * 32 + lane;
    float v = row[c];
    ss += v * att_s[c];
    sd += v * att_d[c];
  }
#pragma unroll
  for (int off = 16; off > 0; off >>= 1) {
    ss += __shfl_xor(ss, off, 32);
    sd += __shfl_xor(sd, off, 32);
  }
  if (lane == 0) { a_s[gw] = ss; a_d[gw] = sd; }
}

__global__ void k_edge_max(const int* __restrict__ src, const int* __restrict__ dst,
                           const float* __restrict__ a_s, const float* __restrict__ a_d,
                           float* __restrict__ m, int nE, int nTot)
{
  int e = blockIdx.x * blockDim.x + threadIdx.x;
  if (e >= nTot) return;
  int s, d;
  if (e < nE) { s = src[e]; d = dst[e]; } else { s = d = e - nE; }
  atomicMaxF(&m[d], lrelu(a_s[s] + a_d[d]));
}

__global__ void k_edge_expsum(const int* __restrict__ src, const int* __restrict__ dst,
                              const float* __restrict__ a_s, const float* __restrict__ a_d,
                              const float* __restrict__ m, float* __restrict__ exv,
                              float* __restrict__ denom, int nE, int nTot)
{
  int e = blockIdx.x * blockDim.x + threadIdx.x;
  if (e >= nTot) return;
  int s, d;
  if (e < nE) { s = src[e]; d = dst[e]; } else { s = d = e - nE; }
  float ex = expf(lrelu(a_s[s] + a_d[d]) - m[d]);
  exv[e] = ex;
  atomicAdd(&denom[d], ex);
}

__global__ void k_edge_scatter(const int* __restrict__ src, const int* __restrict__ dst,
                               const float* __restrict__ exv, const float* __restrict__ denom,
                               const float* __restrict__ xw, float* __restrict__ outb,
                               int nE, int nTot)
{
  int gw = (blockIdx.x * blockDim.x + threadIdx.x) >> 5;
  int lane = threadIdx.x & 31;
  if (gw >= nTot) return;
  int s, d;
  if (gw < nE) { s = src[gw]; d = dst[gw]; } else { s = d = gw - nE; }
  float alpha = exv[gw] / (denom[d] + 1e-16f);
  const float* xr = xw + (long)s * DIM;
  float* orow = outb + (long)d * DIM;
#pragma unroll
  for (int i = 0; i < 4; ++i) {
    int c = i * 32 + lane;
    atomicAdd(&orow[c], alpha * xr[c]);
  }
}

__global__ void k_layernorm(float* __restrict__ x, const float* __restrict__ cb,
                            const float* __restrict__ gamma, const float* __restrict__ beta, int M)
{
  int gw = (blockIdx.x * blockDim.x + threadIdx.x) >> 5;
  int lane = threadIdx.x & 31;
  if (gw >= M) return;
  float* row = x + (long)gw * DIM;
  float v[4];
  float s = 0.f;
#pragma unroll
  for (int i = 0; i < 4; ++i) {
    int c = i * 32 + lane;
    v[i] = row[c] + cb[c];
    s += v[i];
  }
#pragma unroll
  for (int off = 16; off > 0; off >>= 1) s += __shfl_xor(s, off, 32);
  float mu = s * (1.f / DIM);
  float var = 0.f;
#pragma unroll
  for (int i = 0; i < 4; ++i) { float d0 = v[i] - mu; var += d0 * d0; }
#pragma unroll
  for (int off = 16; off > 0; off >>= 1) var += __shfl_xor(var, off, 32);
  float inv = rsqrtf(var * (1.f / DIM) + LN_EPS);
#pragma unroll
  for (int i = 0; i < 4; ++i) {
    int c = i * 32 + lane;
    row[c] = (v[i] - mu) * inv * gamma[c] + beta[c];
  }
}

__global__ void k_scatter_msg(const int* __restrict__ src, const int* __restrict__ dst,
                              const float* __restrict__ z, float* __restrict__ h1, int nE)
{
  int gw = (blockIdx.x * blockDim.x + threadIdx.x) >> 5;
  int lane = threadIdx.x & 31;
  if (gw >= nE) return;
  const float* zr = z + (long)src[gw] * DIM;
  float* hr = h1 + (long)dst[gw] * DIM;
#pragma unroll
  for (int i = 0; i < 4; ++i) {
    int c = i * 32 + lane;
    atomicAdd(&hr[c], zr[c]);
  }
}

__global__ void k_final(const float* __restrict__ h2, const float* __restrict__ w3,
                        const float* __restrict__ b3, float* __restrict__ out, int M)
{
  int gw = (blockIdx.x * blockDim.x + threadIdx.x) >> 5;
  int lane = threadIdx.x & 31;
  if (gw >= M) return;
  const float* row = h2 + (long)gw * 64;
  float s = row[lane] * w3[lane] + row[lane + 32] * w3[lane + 32];
#pragma unroll
  for (int off = 16; off > 0; off >>= 1) s += __shfl_xor(s, off, 32);
  if (lane == 0) out[gw] = s + b3[0];
}

// ---------------------------------------------------------------- launcher

extern "C" void kernel_launch(void* const* d_in, const int* in_sizes, int n_in,
                              void* d_out, int out_size, void* d_ws, size_t ws_size,
                              hipStream_t stream) {
  const float* tx_x      = (const float*)d_in[0];
  const float* emb       = (const float*)d_in[1];
  const float* lin_w     = (const float*)d_in[2];
  const float* att_src   = (const float*)d_in[3];
  const float* att_dst   = (const float*)d_in[4];
  const float* conv_bias = (const float*)d_in[5];
  const float* ln_gamma  = (const float*)d_in[6];
  const float* ln_beta   = (const float*)d_in[7];
  const float* w1        = (const float*)d_in[8];
  const float* b1        = (const float*)d_in[9];
  const float* w2        = (const float*)d_in[10];
  const float* b2        = (const float*)d_in[11];
  const float* w3        = (const float*)d_in[12];
  const float* b3        = (const float*)d_in[13];
  const int* entity_idx  = (const int*)d_in[14];
  const int* edge_src    = (const int*)d_in[15];
  const int* edge_dst    = (const int*)d_in[16];
  float* out = (float*)d_out;

  char* ws = (char*)d_ws;
  size_t off = 0;
  auto alloc = [&](size_t bytes) -> void* {
    off = (off + 255) & ~(size_t)255;
    void* p = ws + off;
    off += bytes;
    return p;
  };

  // packed fragment sizes (halves)
  const int LINW_F  = 4 * 32 * DIM;     // K=128, N=128
  const int W1TOP_F = 13 * 32 * DIM;    // K=394 padded to 416, N=128
  const int W1BLK_F = 4 * 32 * DIM;     // K=128, N=128 (per type)
  const int W2_F    = 4 * 32 * 64;      // K=128, N=64

  _Float16* linw_f  = (_Float16*)alloc((size_t)LINW_F * 2);
  _Float16* w1top_f = (_Float16*)alloc((size_t)W1TOP_F * 2);
  _Float16* w1blk_f = (_Float16*)alloc((size_t)W1BLK_F * T_TYPES * 2);
  _Float16* w2_f    = (_Float16*)alloc((size_t)W2_F * 2);
  float* xw    = (float*)alloc((size_t)N_ENT * DIM * 4);   // also reused as z
  float* outb  = (float*)alloc((size_t)N_ENT * DIM * 4);   // segment-sum -> h (in place)
  float* h1    = (float*)alloc((size_t)N_TX * DIM * 4);
  float* h2    = (float*)alloc((size_t)N_TX * 64 * 4);
  float* a_s   = (float*)alloc((size_t)N_ENT * 4);
  float* a_d   = (float*)alloc((size_t)N_ENT * 4);
  float* mbuf  = (float*)alloc((size_t)N_ENT * 4);
  float* denom = (float*)alloc((size_t)N_ENT * 4);
  float* exv   = (float*)alloc((size_t)(E_EDGES + N_ENT) * 4);
  (void)ws_size; (void)in_sizes; (void)n_in; (void)out_size;

  auto cdiv = [](int a, int b) { return (a + b - 1) / b; };
  const int nTot = E_EDGES + N_ENT;
  const int gemmBlocks  = cdiv(N_ENT, BM);
  const int rowWaveBlks = cdiv(N_ENT * 32, 256);
  const int txWaveBlks  = cdiv(N_TX * 32, 256);

  // ---- pack weights into WMMA B-fragment order (once) ----
  k_pack_bfrag<<<cdiv(LINW_F, 256), 256, 0, stream>>>(lin_w, linw_f, DIM, DIM, LINW_F);
  k_pack_bfrag<<<cdiv(W1TOP_F, 256), 256, 0, stream>>>(w1, w1top_f, F_TX, DIM, W1TOP_F);
  for (int t = 0; t < T_TYPES; ++t)
    k_pack_bfrag<<<cdiv(W1BLK_F, 256), 256, 0, stream>>>(
        w1 + (long)(F_TX + t * DIM) * DIM, w1blk_f + (long)t * W1BLK_F, DIM, DIM, W1BLK_F);
  k_pack_bfrag<<<cdiv(W2_F, 256), 256, 0, stream>>>(w2, w2_f, DIM, 64, W2_F);

  // ---- h1 = tx_x @ w1[:F] + b1  (scatter-adds accumulate on top) ----
  k_gemm_wmma<8><<<cdiv(N_TX, BM), 256, 0, stream>>>(
      tx_x, nullptr, w1top_f, h1, b1, N_TX, F_TX, F_TX, 0);

  // ---- per entity type: GAT conv + message GEMM + scatter ----
  for (int t = 0; t < T_TYPES; ++t) {
    const int* es = edge_src + (long)t * E_EDGES;
    const int* ed = edge_dst + (long)t * E_EDGES;

    k_fill<<<cdiv(N_ENT, 256), 256, 0, stream>>>(mbuf, -1e30f, N_ENT);
    k_fill<<<cdiv(N_ENT, 256), 256, 0, stream>>>(denom, 0.f, N_ENT);
    k_fill<<<cdiv(N_ENT * DIM, 256), 256, 0, stream>>>(outb, 0.f, N_ENT * DIM);

    // xw = emb[t][entity_idx[t]] @ lin_w   (gather fused into WMMA GEMM)
    k_gemm_wmma<8><<<gemmBlocks, 256, 0, stream>>>(
        emb + (long)t * N_ENT * DIM, entity_idx + (long)t * N_ENT,
        linw_f, xw, nullptr, N_ENT, DIM, DIM, 0);

    k_att_dots<<<rowWaveBlks, 256, 0, stream>>>(xw, att_src, att_dst, a_s, a_d, N_ENT);

    k_edge_max<<<cdiv(nTot, 256), 256, 0, stream>>>(es, ed, a_s, a_d, mbuf, E_EDGES, nTot);
    k_edge_expsum<<<cdiv(nTot, 256), 256, 0, stream>>>(es, ed, a_s, a_d, mbuf, exv, denom,
                                                       E_EDGES, nTot);
    k_edge_scatter<<<cdiv(nTot * 32, 256), 256, 0, stream>>>(es, ed, exv, denom, xw, outb,
                                                             E_EDGES, nTot);

    k_layernorm<<<rowWaveBlks, 256, 0, stream>>>(outb, conv_bias,
                                                 ln_gamma + t * DIM, ln_beta + t * DIM, N_ENT);

    // z = h @ w1_block_t  (reuse xw buffer; scatter-add commutes with matmul)
    k_gemm_wmma<8><<<gemmBlocks, 256, 0, stream>>>(
        outb, nullptr, w1blk_f + (long)t * W1BLK_F, xw, nullptr, N_ENT, DIM, DIM, 0);

    k_scatter_msg<<<cdiv(E_EDGES * 32, 256), 256, 0, stream>>>(es, ed, xw, h1, E_EDGES);
  }

  // ---- classifier tail ----
  k_relu<<<cdiv(N_TX * DIM, 256), 256, 0, stream>>>(h1, N_TX * DIM);
  k_gemm_wmma<4><<<cdiv(N_TX, BM), 256, 0, stream>>>(
      h1, nullptr, w2_f, h2, b2, N_TX, DIM, DIM, 1);
  k_final<<<txWaveBlks, 256, 0, stream>>>(h2, w3, b3, out, N_TX);
}
